// Attention1_89790586290351
// MI455X (gfx1250) — matrix operand order repaired and checked by reading, not compiled, not run on hardware
//
#include <hip/hip_runtime.h>

// Problem constants (from reference): B=16, C=NF=256, H=W=128.
#define C_DIM 256
#define HW_DIM 16384
#define NB 16

#define AS1 __attribute__((address_space(1)))
#define AS3 __attribute__((address_space(3)))

#if defined(__has_builtin)
#if __has_builtin(__builtin_amdgcn_global_load_async_to_lds_b128) && \
    __has_builtin(__builtin_amdgcn_s_wait_asynccnt)
#define HAVE_ASYNC_LDS 1
#endif
#endif
#ifndef HAVE_ASYNC_LDS
#define HAVE_ASYNC_LDS 0
#endif

typedef __attribute__((ext_vector_type(16))) __bf16 v16bf;
typedef __attribute__((ext_vector_type(8)))  float  v8f;
typedef __attribute__((ext_vector_type(4)))  int    v4i;

union Frag { v16bf v; uint4 q[2]; __bf16 e[16]; };
union Acc  { v8f  v; float e[8]; };

// D = A(16x32 bf16) * B(32x16 bf16) + C(16x16 f32)
__device__ __forceinline__ v8f wmma_bf16(const Frag& a, const Frag& b, v8f c) {
    return __builtin_amdgcn_wmma_f32_16x16x32_bf16(
        /*neg_a=*/false, a.v, /*neg_b=*/false, b.v,
        /*c_mod=*/(short)0, c, /*reuse_a=*/false, /*reuse_b=*/false);
}

// ---------------------------------------------------------------- utilities
__global__ void f32_to_bf16_kernel(const float* __restrict__ src,
                                   __bf16* __restrict__ dst, int n) {
    int i = blockIdx.x * blockDim.x + threadIdx.x;
    if (i < n) dst[i] = (__bf16)src[i];
}

__global__ void zero_f32_kernel(float* __restrict__ p, int n) {
    int i = blockIdx.x * blockDim.x + threadIdx.x;
    if (i < n) p[i] = 0.0f;
}

// load 4 consecutive B elements as bf16 (with conversion if f32 source)
__device__ __forceinline__ void load4_bf16(const float* p, __bf16* o) {
    float4 f = *(const float4*)p;
    o[0] = (__bf16)f.x; o[1] = (__bf16)f.y; o[2] = (__bf16)f.z; o[3] = (__bf16)f.w;
}
__device__ __forceinline__ void load4_bf16(const __bf16* p, __bf16* o) {
    uint2 u = *(const uint2*)p;
    union { unsigned int w; __bf16 h[2]; } a, b;
    a.w = u.x; b.w = u.y;
    o[0] = a.h[0]; o[1] = a.h[1]; o[2] = b.h[0]; o[3] = b.h[1];
}

// --------------------------------------------------------------- main GEMM
// Y[b, m, n] = sum_k A[(b,)m, k] * B[b, k, n] (+ bias[m])
// A: bf16 row-major [256 x 256]   (weights, or per-batch attention)
// B: TB  row-major [B x 256 x HW]
// WG = 256 threads = 8 waves; WG tile 64(M) x 128(N); wave tile 32x32 (2x2 WMMA)
template <typename TB, typename TOUT, bool HAS_BIAS, bool A_BATCHED>
__global__ __launch_bounds__(256) void gemm_wmma_kernel(
    const __bf16* __restrict__ Ag,
    const TB*     __restrict__ Bg,
    const float*  __restrict__ bias,
    TOUT*         __restrict__ Yg) {
    __shared__ __align__(16) __bf16 lA[64][40];    // [m][k], padded stride
    __shared__ __align__(16) __bf16 lB[128][40];   // [n][k] (transposed stage)

    const int tid  = threadIdx.x;
    const int wave = tid >> 5, lane = tid & 31;
    const int lm   = lane & 15, half = lane >> 4;
    const int wm   = wave >> 2, wn = wave & 3;     // 2 x 4 wave grid

    const int b  = blockIdx.z;
    const int m0 = blockIdx.y * 64;
    const int n0 = blockIdx.x * 128;

    const __bf16* A  = Ag + (A_BATCHED ? (size_t)b * C_DIM * C_DIM : 0);
    const TB*     Bb = Bg + (size_t)b * C_DIM * HW_DIM;
    TOUT*         Y  = Yg + (size_t)b * C_DIM * HW_DIM;

    Acc acc[2][2];
    for (int mi = 0; mi < 2; ++mi)
        for (int ni = 0; ni < 2; ++ni)
            for (int i = 0; i < 8; ++i) acc[mi][ni].e[i] = 0.0f;

    for (int k0 = 0; k0 < C_DIM; k0 += 32) {
        // ---- stage A tile 64x32 (K-contiguous). One b128 per thread;
        //      use the gfx1250 async global->LDS path when available.
        {
            const int r  = tid >> 2;           // 0..63
            const int kc = (tid & 3) * 8;      // 0,8,16,24
            const __bf16* gsrc = A + (size_t)(m0 + r) * C_DIM + k0 + kc;
#if HAVE_ASYNC_LDS
            __builtin_amdgcn_global_load_async_to_lds_b128(
                (AS1 v4i*)gsrc, (AS3 v4i*)&lA[r][kc], 0, 0);
#else
            *(uint4*)&lA[r][kc] = *(const uint4*)gsrc;
#endif
            if (k0 + 32 < C_DIM)
                __builtin_prefetch(gsrc + 32, 0, 3);
        }
        // ---- stage B tile 32(K) x 128(N), transposed into lB[n][k].
        //      Each thread owns a K-row *pair* so transposed stores are b32.
        {
            const int cb = (tid & 31) * 4;     // col 0..124
            const int rp = (tid >> 5) * 2;     // row-pair base 0..14
            for (int i = 0; i < 2; ++i) {
                const int r = rp + i * 16;     // rows r, r+1 in 0..31
                __bf16 t0[4], t1[4];
                const TB* s0 = Bb + (size_t)(k0 + r) * HW_DIM + n0 + cb;
                const TB* s1 = s0 + HW_DIM;
                load4_bf16(s0, t0);
                load4_bf16(s1, t1);
                if (k0 + 32 < C_DIM)
                    __builtin_prefetch(s0 + 32 * HW_DIM, 0, 3);
                for (int j = 0; j < 4; ++j) {
                    union { unsigned int w; __bf16 h[2]; } p;
                    p.h[0] = t0[j]; p.h[1] = t1[j];
                    *(unsigned int*)&lB[cb + j][r] = p.w;
                }
            }
        }
#if HAVE_ASYNC_LDS
        __builtin_amdgcn_s_wait_asynccnt(0);
#endif
        __syncthreads();

        // ---- fragments (ISA 7.12.2 layouts) + 2x2 WMMA
        Frag af[2], bfr[2];
        for (int mi = 0; mi < 2; ++mi) {
            const int mr = wm * 32 + mi * 16 + lm;
            af[mi].q[0] = *(const uint4*)&lA[mr][half * 8];       // K 0..7 / 8..15
            af[mi].q[1] = *(const uint4*)&lA[mr][16 + half * 8];  // K 16..23 / 24..31
        }
        for (int ni = 0; ni < 2; ++ni) {
            const int nr = wn * 32 + ni * 16 + lm;
            bfr[ni].q[0] = *(const uint4*)&lB[nr][half * 8];       // K 0..7 / 8..15
            bfr[ni].q[1] = *(const uint4*)&lB[nr][16 + half * 8];  // K 16..23 / 24..31
        }
        for (int mi = 0; mi < 2; ++mi)
            for (int ni = 0; ni < 2; ++ni)
                acc[mi][ni].v = wmma_bf16(af[mi], bfr[ni], acc[mi][ni].v);
        __syncthreads();
    }

    // ---- epilogue: C/D layout row = 8*half + i, col = lm
    for (int mi = 0; mi < 2; ++mi)
        for (int ni = 0; ni < 2; ++ni)
            for (int i = 0; i < 8; ++i) {
                const int m = m0 + wm * 32 + mi * 16 + half * 8 + i;
                const int n = n0 + wn * 32 + ni * 16 + lm;
                float v = acc[mi][ni].e[i];
                if (HAS_BIAS) v += bias[m];
                Y[(size_t)m * HW_DIM + n] = (TOUT)v;
            }
}

// --------------------------------------------------------- energy (q k^T)
// energy[b, c, d] += sum over an HW chunk; A and B both K-contiguous bf16,
// so fragments load straight from global (no LDS). K split over gridDim.z.
__global__ __launch_bounds__(256) void energy_wmma_kernel(
    const __bf16* __restrict__ qb, const __bf16* __restrict__ kb,
    float* __restrict__ energy) {
    const int tid  = threadIdx.x;
    const int wave = tid >> 5, lane = tid & 31;
    const int lm   = lane & 15, half = lane >> 4;
    const int b    = blockIdx.y;
    const int tile = blockIdx.x * 8 + wave;        // 0..255 -> 16x16 tiles
    const int tm = tile >> 4, tn = tile & 15;

    const size_t base = (size_t)b * C_DIM * HW_DIM;
    const __bf16* qr = qb + base + (size_t)(tm * 16 + lm) * HW_DIM;
    const __bf16* kr = kb + base + (size_t)(tn * 16 + lm) * HW_DIM;

    Acc acc;
    for (int i = 0; i < 8; ++i) acc.e[i] = 0.0f;

    const int k0 = blockIdx.z * 2048;
    for (int kk = k0; kk < k0 + 2048; kk += 32) {
        __builtin_prefetch(qr + kk + 256, 0, 3);   // global_prefetch_b8
        __builtin_prefetch(kr + kk + 256, 0, 3);
        Frag a, bf_;
        a.q[0]   = *(const uint4*)(qr + kk + half * 8);
        a.q[1]   = *(const uint4*)(qr + kk + 16 + half * 8);
        bf_.q[0] = *(const uint4*)(kr + kk + half * 8);
        bf_.q[1] = *(const uint4*)(kr + kk + 16 + half * 8);
        acc.v = wmma_bf16(a, bf_, acc.v);
    }

    float* e = energy + (size_t)b * C_DIM * C_DIM;
    for (int i = 0; i < 8; ++i)
        atomicAdd(&e[(size_t)(tm * 16 + half * 8 + i) * C_DIM + tn * 16 + lm],
                  acc.e[i]);
}

// ------------------------------------------------- negated softmax per row
// softmax(rowmax - e) == exp(rowmin - e) / sum(exp(rowmin - e))
__global__ __launch_bounds__(256) void softmax_kernel(
    const float* __restrict__ energy, __bf16* __restrict__ attn) {
    const int row = blockIdx.x;                     // b*256 + c
    const int tid = threadIdx.x;
    const float e = energy[(size_t)row * C_DIM + tid];

    __shared__ float red[8];
    __shared__ float bcast;

    float v = e;
    for (int o = 16; o; o >>= 1) v = fminf(v, __shfl_xor(v, o, 32));
    if ((tid & 31) == 0) red[tid >> 5] = v;
    __syncthreads();
    if (tid == 0) {
        float m = red[0];
        for (int i = 1; i < 8; ++i) m = fminf(m, red[i]);
        bcast = m;
    }
    __syncthreads();
    const float rmin = bcast;
    const float ex = __expf(rmin - e);

    v = ex;
    for (int o = 16; o; o >>= 1) v += __shfl_xor(v, o, 32);
    __syncthreads();
    if ((tid & 31) == 0) red[tid >> 5] = v;
    __syncthreads();
    if (tid == 0) {
        float s = 0.0f;
        for (int i = 0; i < 8; ++i) s += red[i];
        bcast = s;
    }
    __syncthreads();
    attn[(size_t)row * C_DIM + tid] = (__bf16)(ex / bcast);
}

// ---------------------------------------------------------------- launcher
extern "C" void kernel_launch(void* const* d_in, const int* in_sizes, int n_in,
                              void* d_out, int out_size, void* d_ws, size_t ws_size,
                              hipStream_t stream) {
    (void)in_sizes; (void)n_in; (void)out_size; (void)ws_size;
    const float* query = (const float*)d_in[0];
    const float* key_  = (const float*)d_in[1];
    const float* x     = (const float*)d_in[2];
    const float* Wq    = (const float*)d_in[3];
    const float* bq    = (const float*)d_in[4];
    const float* Wk    = (const float*)d_in[5];
    const float* bk    = (const float*)d_in[6];
    const float* Wv    = (const float*)d_in[7];
    const float* bv    = (const float*)d_in[8];
    float* out = (float*)d_out;

    char* ws = (char*)d_ws;
    const size_t qkv_bytes = (size_t)NB * C_DIM * HW_DIM * sizeof(__bf16); // 128 MiB
    __bf16* qb  = (__bf16*)ws; ws += qkv_bytes;
    __bf16* kb  = (__bf16*)ws; ws += qkv_bytes;
    __bf16* vb  = (__bf16*)ws; ws += qkv_bytes;
    __bf16* Wqb = (__bf16*)ws; ws += (size_t)C_DIM * C_DIM * sizeof(__bf16);
    __bf16* Wkb = (__bf16*)ws; ws += (size_t)C_DIM * C_DIM * sizeof(__bf16);
    __bf16* Wvb = (__bf16*)ws; ws += (size_t)C_DIM * C_DIM * sizeof(__bf16);
    float* energy = (float*)ws; ws += (size_t)NB * C_DIM * C_DIM * sizeof(float);
    __bf16* attn  = (__bf16*)ws; ws += (size_t)NB * C_DIM * C_DIM * sizeof(__bf16);

    // 1) weights -> bf16
    f32_to_bf16_kernel<<<C_DIM * C_DIM / 256, 256, 0, stream>>>(Wq, Wqb, C_DIM * C_DIM);
    f32_to_bf16_kernel<<<C_DIM * C_DIM / 256, 256, 0, stream>>>(Wk, Wkb, C_DIM * C_DIM);
    f32_to_bf16_kernel<<<C_DIM * C_DIM / 256, 256, 0, stream>>>(Wv, Wvb, C_DIM * C_DIM);

    // 2) 1x1 convs (q,k,v) -> bf16 intermediates
    dim3 gg(HW_DIM / 128, C_DIM / 64, NB);   // (128, 4, 16)
    gemm_wmma_kernel<float, __bf16, true, false><<<gg, 256, 0, stream>>>(Wqb, query, bq, qb);
    gemm_wmma_kernel<float, __bf16, true, false><<<gg, 256, 0, stream>>>(Wkb, key_,  bk, kb);
    gemm_wmma_kernel<float, __bf16, true, false><<<gg, 256, 0, stream>>>(Wvb, x,     bv, vb);

    // 3) energy = q k^T  (zero, then K-split accumulate with f32 atomics)
    const int e_n = NB * C_DIM * C_DIM;
    zero_f32_kernel<<<(e_n + 255) / 256, 256, 0, stream>>>(energy, e_n);
    energy_wmma_kernel<<<dim3(32, NB, 8), 256, 0, stream>>>(qb, kb, energy);

    // 4) negated softmax -> bf16 attention
    softmax_kernel<<<NB * C_DIM, 256, 0, stream>>>(energy, attn);

    // 5) out = attention @ v  (fp32 output)
    gemm_wmma_kernel<__bf16, float, false, true><<<gg, 256, 0, stream>>>(attn, vb, nullptr, out);
}
